// LSTM_40484361732221
// MI455X (gfx1250) — matrix operand order repaired and checked by reading, not compiled, not run on hardware
//
#include <hip/hip_runtime.h>
#include <hip/hip_bf16.h>

typedef __attribute__((ext_vector_type(16))) _Float16 v16h;
typedef __attribute__((ext_vector_type(8)))  float    v8f;

#define BROWS   16    // batch rows per block
#define KTOT    96    // 64 (h) + 8 (x) + 24 (zero pad) -> 3 K-chunks of 32
#define T_STEPS 512
#define TCHUNK  128   // output staging chunk in LDS

#if __has_builtin(__builtin_amdgcn_tanhf)
#define TANHF(x) __builtin_amdgcn_tanhf(x)
#else
#define TANHF(x) tanhf(x)
#endif

__device__ __forceinline__ float fsigmoid(float x) {
    // sigmoid(x) = 0.5*tanh(x/2) + 0.5  -> v_tanh_f32 fast path
    return 0.5f * TANHF(0.5f * x) + 0.5f;
}

union AFrag {
    v16h v;
    unsigned int u[8];
    _Float16 h[16];
};

// Concatenated weight matrix Wcat[k][n]:
//   k in [0,64):  W_hh[n][k]    (h contribution)
//   k in [64,72): W_ih[n][k-64] (x contribution)
//   k in [72,96): 0             (pad)
__device__ __forceinline__ float wcat(const float* __restrict__ whh,
                                      const float* __restrict__ wih,
                                      int k, int n) {
    if (k < 64) return whh[n * 64 + k];
    if (k < 72) return wih[n * 8 + (k - 64)];
    return 0.0f;
}

__global__ __launch_bounds__(128, 1)
void lstm_fused_wmma(const float* __restrict__ x,    // [4096,512,8]
                     const float* __restrict__ Wih,  // [256,8]
                     const float* __restrict__ Whh,  // [256,64]
                     const float* __restrict__ bih,  // [256]
                     const float* __restrict__ bhh,  // [256]
                     const float* __restrict__ Wd,   // [64]
                     const float* __restrict__ bd,   // [1]
                     float* __restrict__ out)        // [4096,512]
{
    __shared__ _Float16 hA[2][BROWS][KTOT];     // double-buffered [h|x|pad] A-tile
    __shared__ float outBuf[BROWS][TCHUNK];     // dense-head staging

    const int tid  = threadIdx.x;
    const int lane = tid & 31;
    const int w    = tid >> 5;     // wave id 0..3 -> hidden cols [16w,16w+16)
    const int lo   = lane & 15;
    const int hi   = lane >> 4;    // high half-wave selector
    const int b0   = blockIdx.x * BROWS;

    // ---- one-time setup: B-matrix fragments + biases pinned in VGPRs ----
    // B layout (f16, 32x16 per chunk): VGPR j: K = 8*hi + (j<4?0:16) + 2(j&3) + {0,1};
    // matches A fragment order; N = lane%16.
    AFrag bf[4][3];
    float bias[4];
#pragma unroll
    for (int g = 0; g < 4; ++g) {
        const int n = g * 64 + w * 16 + lo;   // gate-major column index
        bias[g] = bih[n] + bhh[n];
#pragma unroll
        for (int kc = 0; kc < 3; ++kc) {
#pragma unroll
            for (int j = 0; j < 8; ++j) {
                const int k = kc * 32 + hi * 8 + ((j < 4) ? 0 : 16) + 2 * (j & 3);
                bf[g][kc].h[2 * j]     = (_Float16)wcat(Whh, Wih, k, n);
                bf[g][kc].h[2 * j + 1] = (_Float16)wcat(Whh, Wih, k + 1, n);
            }
        }
    }

    // W_dense packed f16 to mirror the A-fragment half layout (K = 0..63):
    // A half q (q=2j+h') of chunk kc holds K = kc*32 + hi*8 + (j<4?0:16) + 2(j&3) + h'
    AFrag wdh[2];
#pragma unroll
    for (int kc = 0; kc < 2; ++kc) {
#pragma unroll
        for (int q = 0; q < 16; ++q) {
            const int j = q >> 1, hp = q & 1;
            const int K = kc * 32 + hi * 8 + ((j < 4) ? 0 : 16) + 2 * (j & 3) + hp;
            wdh[kc].h[q] = (_Float16)Wd[K];
        }
    }
    const float bdv = bd[0];

    // ---- init LDS: zero A-tiles (h0 = 0, pad = 0) ----
    for (int idx = tid; idx < 2 * BROWS * KTOT; idx += 128)
        ((_Float16*)hA)[idx] = (_Float16)0.0f;

    const int xrow = tid >> 3, xi = tid & 7;   // 128 threads <-> 16x8 x-tile
    hA[0][xrow][64 + xi] =
        (_Float16)x[((size_t)(b0 + xrow) * T_STEPS + 0) * 8 + xi];
    __syncthreads();

    v8f c = {};   // cell-state fragment (C/D layout): VGPR r -> M=r+8*hi, N=lo

    for (int t = 0; t < T_STEPS; ++t) {
        const int cur = t & 1, nxt = cur ^ 1;

        // ---- flush a completed 128-step output chunk (coalesced float4) ----
        if (t > 0 && (t & (TCHUNK - 1)) == 0) {
            const int tbase = t - TCHUNK;
            for (int idx = tid * 4; idx < BROWS * TCHUNK; idx += 128 * 4) {
                const int row = idx / TCHUNK, tt = idx % TCHUNK;
                *(float4*)&out[(size_t)(b0 + row) * T_STEPS + tbase + tt] =
                    *(const float4*)&outBuf[row][tt];
            }
            __syncthreads();
        }

        // prefetch x_{t+1} (L2-resident) early; consumed after WMMAs
        float xv = 0.0f;
        if (t + 1 < T_STEPS)
            xv = x[((size_t)(b0 + xrow) * T_STEPS + (t + 1)) * 8 + xi];

        // ---- gather A fragments from LDS per documented 16-bit A layout ----
        AFrag a[3];
#pragma unroll
        for (int kc = 0; kc < 3; ++kc) {
#pragma unroll
            for (int j = 0; j < 8; ++j) {
                const int koff = kc * 32 + (hi ? 8 : 0) + ((j < 4) ? 0 : 16) + 2 * (j & 3);
                a[kc].u[j] = *(const unsigned int*)&hA[cur][lo][koff];
            }
        }

        // ---- 12 WMMAs: 4 gates x 3 K-chunks, f32 accumulate ----
        v8f acc[4];
#pragma unroll
        for (int g = 0; g < 4; ++g) {
            v8f z = {};
            acc[g] = __builtin_amdgcn_wmma_f32_16x16x32_f16(false, a[0].v, false, bf[g][0].v, (short)0, z,      false, false);
            acc[g] = __builtin_amdgcn_wmma_f32_16x16x32_f16(false, a[1].v, false, bf[g][1].v, (short)0, acc[g], false, false);
            acc[g] = __builtin_amdgcn_wmma_f32_16x16x32_f16(false, a[2].v, false, bf[g][2].v, (short)0, acc[g], false, false);
        }

        // ---- dense head for out[:, t-1] from the A fragments (h_{t-1}) ----
        // Lane (lo,hi) holds 32 of the 64 h values of batch row lo; the other
        // half-wave holds the rest. 32 mixed f16*f16+f32 FMAs + one shfl_xor(16).
        if (w == 0 && t > 0) {
            float s = 0.0f;
#pragma unroll
            for (int kc = 0; kc < 2; ++kc)
#pragma unroll
                for (int q = 0; q < 16; ++q)
                    s += (float)a[kc].h[q] * (float)wdh[kc].h[q];
            s += __shfl_xor(s, 16, 32);
            if (hi == 0)
                outBuf[lo][(t - 1) & (TCHUNK - 1)] = s + bdv;
        }

        // ---- LSTM cell update + h write-back ----
        const int jcol = w * 16 + lo;
#pragma unroll
        for (int r = 0; r < 8; ++r) {
            const float ig = fsigmoid(acc[0][r] + bias[0]);
            const float fg = fsigmoid(acc[1][r] + bias[1]);
            const float gg = TANHF(acc[2][r] + bias[2]);
            const float og = fsigmoid(acc[3][r] + bias[3]);
            const float cv = fg * c[r] + ig * gg;
            c[r] = cv;
            const float hv = og * TANHF(cv);
            const int mrow = r + (hi ? 8 : 0);
            hA[nxt][mrow][jcol] = (_Float16)hv;      // h_t for next step's A
        }

        // stage x_{t+1} into the next A-tile
        if (t + 1 < T_STEPS)
            hA[nxt][xrow][64 + xi] = (_Float16)xv;

        __syncthreads();
    }

    // ---- epilogue: out[:, 511] from the final h tile (in hA[0]) ----
    if (tid < BROWS) {
        float s = bdv;
        for (int j = 0; j < 64; ++j)
            s += (float)hA[0][tid][j] * Wd[j];
        outBuf[tid][TCHUNK - 1] = s;
    }
    __syncthreads();

    // ---- flush final chunk (t = 384..511) ----
    {
        const int tbase = T_STEPS - TCHUNK;
        for (int idx = tid * 4; idx < BROWS * TCHUNK; idx += 128 * 4) {
            const int row = idx / TCHUNK, tt = idx % TCHUNK;
            *(float4*)&out[(size_t)(b0 + row) * T_STEPS + tbase + tt] =
                *(const float4*)&outBuf[row][tt];
        }
    }
}

extern "C" void kernel_launch(void* const* d_in, const int* in_sizes, int n_in,
                              void* d_out, int out_size, void* d_ws, size_t ws_size,
                              hipStream_t stream) {
    const float* x   = (const float*)d_in[0];
    const float* Wih = (const float*)d_in[1];
    const float* Whh = (const float*)d_in[2];
    const float* bih = (const float*)d_in[3];
    const float* bhh = (const float*)d_in[4];
    const float* Wd  = (const float*)d_in[5];
    const float* bd  = (const float*)d_in[6];
    float* out = (float*)d_out;

    dim3 grid(4096 / BROWS);   // 256 blocks, one 16-row batch tile each
    dim3 block(128);           // 4 wave32 waves
    lstm_fused_wmma<<<grid, block, 0, stream>>>(x, Wih, Whh, bih, bhh, Wd, bd, out);
}
